// Critic_QuadNeighborhood_MultiheadAttention_80169859547822
// MI455X (gfx1250) — matrix operand order repaired and checked by reading, not compiled, not run on hardware
//
#include <hip/hip_runtime.h>
#include <hip/hip_bf16.h>

#define BSZ      16384
#define NNEI     7
#define HD       512
#define MROWS    (BSZ * NNEI)
#define NGRP     (BSZ / 8)
#define OUT1_ELEMS ((size_t)BSZ * HD)

typedef __attribute__((ext_vector_type(16))) __bf16 v16bf;
typedef __attribute__((ext_vector_type(8)))  __bf16 v8bf;
typedef __attribute__((ext_vector_type(8)))  float  v8f;

__device__ __forceinline__ unsigned short f2bfu(float f) {
    union { float f; unsigned int u; } x; x.f = f;
    unsigned int u = x.u + 0x7FFFu + ((x.u >> 16) & 1u);
    return (unsigned short)(u >> 16);
}

__device__ __forceinline__ float bf2f(unsigned short s) {
    union { unsigned int u; float f; } x; x.u = ((unsigned int)s) << 16;
    return x.f;
}

__device__ __forceinline__ float fast_tanh(float x) {
#if __has_builtin(__builtin_amdgcn_tanhf)
    return __builtin_amdgcn_tanhf(x);
#else
    return tanhf(x);
#endif
}

// ---------------------------------------------------------------------------
// Pack fp32 weight (Krows x 512 row-major) into WMMA-B bf16 layout.
// ---------------------------------------------------------------------------
__global__ void pack_B_kernel(const float* __restrict__ W,
                              unsigned short* __restrict__ out,
                              int Krows, int total) {
    int o = blockIdx.x * 256 + threadIdx.x;
    if (o >= total) return;
    int e    = o & 15;
    int lane = (o >> 4) & 31;
    int n16  = (o >> 9) & 31;
    int kt   = o >> 14;
    int kl   = ((e & 8) << 1) + ((lane >> 4) << 3) + (e & 7);
    int k    = kt * 32 + kl;
    int n    = (n16 << 4) + (lane & 15);
    float v  = (k < Krows) ? W[(size_t)k * 512 + n] : 0.0f;
    out[o] = f2bfu(v);
}

// ---------------------------------------------------------------------------
// Build stage-1 input X (MROWS x 32 bf16), K padded 24->32 with zeros.
// ---------------------------------------------------------------------------
__global__ void build_X_kernel(const float* __restrict__ self_obs,
                               const float* __restrict__ obs,
                               unsigned short* __restrict__ X) {
    int idx = blockIdx.x * 256 + threadIdx.x;
    if (idx >= MROWS * 32) return;
    int j = idx >> 5;
    int c = idx & 31;
    float v = 0.0f;
    if (c < 18)       v = self_obs[(size_t)(j % BSZ) * 18 + c];
    else if (c < 24)  v = obs[(size_t)(j / NNEI) * 60 + 18 + (j % NNEI) * 6 + (c - 18)];
    X[idx] = f2bfu(v);
}

// In-flight B slab in 8 named uint4 (32 VGPRs, no array => nothing to SROA)
#define LOAD_BSLAB(Wp, kt)                                              \
    {                                                                   \
        const uint4* bg = (const uint4*)((Wp) + (size_t)(kt) * 16384);  \
        r0 = bg[tid];           r1 = bg[tid + 256];                     \
        r2 = bg[tid + 512];     r3 = bg[tid + 768];                     \
        r4 = bg[tid + 1024];    r5 = bg[tid + 1280];                    \
        r6 = bg[tid + 1536];    r7 = bg[tid + 1792];                    \
    }

#define STORE_BSLAB()                                                   \
    {                                                                   \
        uint4* bl = (uint4*)sB;                                         \
        bl[tid] = r0;           bl[tid + 256] = r1;                     \
        bl[tid + 512] = r2;     bl[tid + 768] = r3;                     \
        bl[tid + 1024] = r4;    bl[tid + 1280] = r5;                    \
        bl[tid + 1536] = r6;    bl[tid + 1792] = r7;                    \
    }

// ---------------------------------------------------------------------------
// Fused 2-layer MLP: C = tanh(tanh(A @ W1 + bias1) @ W2 + bias2)
// 64-row x 512-col block tile, hidden layer resident in LDS.
// ---------------------------------------------------------------------------
__global__ __launch_bounds__(256)
__attribute__((amdgpu_waves_per_eu(1, 4)))
void mlp2_bf16(
    const unsigned short* __restrict__ A, int K1,
    const unsigned short* __restrict__ W1, const float* __restrict__ bias1,
    const unsigned short* __restrict__ W2, const float* __restrict__ bias2,
    unsigned short* __restrict__ C) {

    __shared__ unsigned short sA[64 * 40];
    __shared__ unsigned short sB[32 * 32 * 16];
    __shared__ unsigned short sY[64 * 520];

    const int tid   = threadIdx.x;
    const int lane  = tid & 31;
    const int wave  = tid >> 5;
    const int waveM = wave & 1;
    const int waveN = wave >> 1;
    const size_t mb = (size_t)blockIdx.x * 64;

    const int koff = (lane >> 4) << 3;
    const int rA = tid >> 2, cA = tid & 3;

    v8f acc[2][8];
    v8f vz = {};
    #pragma unroll
    for (int i = 0; i < 2; ++i)
        #pragma unroll
        for (int j = 0; j < 8; ++j) acc[i][j] = vz;

    uint4 r0, r1, r2, r3, r4, r5, r6, r7;
    uint4 aReg;

    // ---------------- Layer 1 ----------------
    const int nKt = K1 >> 5;
    LOAD_BSLAB(W1, 0);
    aReg = *(const uint4*)(A + (mb + rA) * (size_t)K1 + cA * 8);

    for (int kt = 0; kt < nKt; ++kt) {
        *(uint4*)(&sA[rA * 40 + cA * 8]) = aReg;
        STORE_BSLAB();
        __syncthreads();

        if (kt + 1 < nKt) {
            LOAD_BSLAB(W1, kt + 1);
            aReg = *(const uint4*)(A + (mb + rA) * (size_t)K1 + (kt + 1) * 32 + cA * 8);
        }
        if (kt + 2 < nKt) {
            __builtin_prefetch(W1 + (size_t)(kt + 2) * 16384 + tid * 8, 0, 1);
            __builtin_prefetch(A + (mb + rA) * (size_t)K1 + (kt + 2) * 32 + cA * 8, 0, 1);
        }

        v16bf afrag[2];
        #pragma unroll
        for (int mt = 0; mt < 2; ++mt) {
            int row = waveM * 32 + mt * 16 + (lane & 15);
            v8bf lo = *(const v8bf*)(&sA[row * 40 + koff]);
            v8bf hi = *(const v8bf*)(&sA[row * 40 + 16 + koff]);
            afrag[mt] = __builtin_shufflevector(lo, hi,
                0,1,2,3,4,5,6,7,8,9,10,11,12,13,14,15);
        }
        #pragma unroll
        for (int nt = 0; nt < 8; ++nt) {
            int n16 = waveN * 8 + nt;
            v16bf bfrag = *(const v16bf*)(&sB[(n16 * 32 + lane) * 16]);
            acc[0][nt] = __builtin_amdgcn_wmma_f32_16x16x32_bf16(
                false, afrag[0], false, bfrag, (short)0, acc[0][nt], false, false);
            acc[1][nt] = __builtin_amdgcn_wmma_f32_16x16x32_bf16(
                false, afrag[1], false, bfrag, (short)0, acc[1][nt], false, false);
        }
        __syncthreads();
    }

    // Preload W2 slab 0; overlaps with the VALU-heavy epilogue 1.
    LOAD_BSLAB(W2, 0);

    // Epilogue 1: bias + tanh -> LDS hidden
    #pragma unroll
    for (int nt = 0; nt < 8; ++nt) {
        int col = waveN * 128 + nt * 16 + (lane & 15);
        float bc = bias1[col];
        #pragma unroll
        for (int mt = 0; mt < 2; ++mt)
            #pragma unroll
            for (int r = 0; r < 8; ++r) {
                int rowl = waveM * 32 + mt * 16 + r + ((lane >> 4) << 3);
                sY[rowl * 520 + col] = f2bfu(fast_tanh(acc[mt][nt][r] + bc));
            }
    }
    __syncthreads();

    // ---------------- Layer 2 (K = 512, A from LDS) ----------------
    #pragma unroll
    for (int i = 0; i < 2; ++i)
        #pragma unroll
        for (int j = 0; j < 8; ++j) acc[i][j] = vz;

    for (int kt = 0; kt < 16; ++kt) {
        STORE_BSLAB();
        __syncthreads();

        if (kt + 1 < 16) {
            LOAD_BSLAB(W2, kt + 1);
        }
        if (kt + 2 < 16) {
            __builtin_prefetch(W2 + (size_t)(kt + 2) * 16384 + tid * 8, 0, 1);
        }

        v16bf afrag[2];
        #pragma unroll
        for (int mt = 0; mt < 2; ++mt) {
            int row  = waveM * 32 + mt * 16 + (lane & 15);
            int base = row * 520 + kt * 32 + koff;
            v8bf lo = *(const v8bf*)(&sY[base]);
            v8bf hi = *(const v8bf*)(&sY[base + 16]);
            afrag[mt] = __builtin_shufflevector(lo, hi,
                0,1,2,3,4,5,6,7,8,9,10,11,12,13,14,15);
        }
        #pragma unroll
        for (int nt = 0; nt < 8; ++nt) {
            int n16 = waveN * 8 + nt;
            v16bf bfrag = *(const v16bf*)(&sB[(n16 * 32 + lane) * 16]);
            acc[0][nt] = __builtin_amdgcn_wmma_f32_16x16x32_bf16(
                false, afrag[0], false, bfrag, (short)0, acc[0][nt], false, false);
            acc[1][nt] = __builtin_amdgcn_wmma_f32_16x16x32_bf16(
                false, afrag[1], false, bfrag, (short)0, acc[1][nt], false, false);
        }
        __syncthreads();
    }

    // Epilogue 2: bias + tanh -> global bf16
    #pragma unroll
    for (int nt = 0; nt < 8; ++nt) {
        int col = waveN * 128 + nt * 16 + (lane & 15);
        float bc = bias2[col];
        #pragma unroll
        for (int mt = 0; mt < 2; ++mt)
            #pragma unroll
            for (int r = 0; r < 8; ++r) {
                size_t rowg = mb + waveM * 32 + mt * 16 + r + ((lane >> 4) << 3);
                C[rowg * 512 + col] = f2bfu(fast_tanh(acc[mt][nt][r] + bc));
            }
    }
}

// ---------------------------------------------------------------------------
// Agent attention (softmax over 7 neighbors), one wave per batch row.
// ---------------------------------------------------------------------------
__global__ __launch_bounds__(256) void agent_attention_kernel(
    const unsigned short* __restrict__ E, const unsigned short* __restrict__ V,
    float* __restrict__ outF, unsigned short* __restrict__ out16) {

    const int lane = threadIdx.x & 31;
    const int wave = threadIdx.x >> 5;
    const int b = blockIdx.x * 8 + wave;
    const size_t base = (size_t)b * NNEI * HD;

    float e[NNEI][16];
    float q[16];
    #pragma unroll
    for (int c = 0; c < 16; ++c) q[c] = 0.0f;
    #pragma unroll
    for (int i = 0; i < NNEI; ++i)
        #pragma unroll
        for (int c = 0; c < 16; ++c) {
            float v = bf2f(E[base + (size_t)i * HD + c * 32 + lane]);
            e[i][c] = v; q[c] += v;
        }
    const float qscale = (1.0f / 7.0f) * 0.04419417382415922f;
    #pragma unroll
    for (int c = 0; c < 16; ++c) q[c] *= qscale;

    float s[NNEI];
    #pragma unroll
    for (int i = 0; i < NNEI; ++i) {
        float p = 0.0f;
        #pragma unroll
        for (int c = 0; c < 16; ++c) p += q[c] * e[i][c];
        #pragma unroll
        for (int off = 16; off > 0; off >>= 1) p += __shfl_xor(p, off, 32);
        s[i] = p;
    }
    float m = s[0];
    #pragma unroll
    for (int i = 1; i < NNEI; ++i) m = fmaxf(m, s[i]);
    float sum = 0.0f;
    #pragma unroll
    for (int i = 0; i < NNEI; ++i) { s[i] = __expf(s[i] - m); sum += s[i]; }
    float inv = 1.0f / sum;

    #pragma unroll
    for (int c = 0; c < 16; ++c) {
        float o = 0.0f;
        #pragma unroll
        for (int i = 0; i < NNEI; ++i)
            o += s[i] * bf2f(V[base + (size_t)i * HD + c * 32 + lane]);
        o *= inv;
        int d = c * 32 + lane;
        outF[(size_t)b * HD + d]  = o;
        out16[(size_t)b * HD + d] = f2bfu(o);
    }
}

// ---------------------------------------------------------------------------
// Multi-head attention (8 agents, 8 heads x 64 dims), one wave per group.
// ---------------------------------------------------------------------------
__global__ __launch_bounds__(256) void multihead_kernel(
    const float* __restrict__ Aatt, const unsigned short* __restrict__ Vals,
    float* __restrict__ out1) {

    const int lane = threadIdx.x & 31;
    const int wave = threadIdx.x >> 5;
    const int g = blockIdx.x * 8 + wave;
    const float* Ar = Aatt + (size_t)g * 8 * HD;
    const unsigned short* Vr = Vals + (size_t)g * 8 * HD;

    #pragma unroll
    for (int h = 0; h < 8; ++h) {
        int d0 = h * 64 + lane * 2;
        float k0[8], k1[8], q0 = 0.0f, q1 = 0.0f;
        #pragma unroll
        for (int a = 0; a < 8; ++a) {
            k0[a] = Ar[a * HD + d0];
            k1[a] = Ar[a * HD + d0 + 1];
            q0 += k0[a]; q1 += k1[a];
        }
        q0 *= (0.125f * 0.125f);
        q1 *= (0.125f * 0.125f);

        float s[8];
        #pragma unroll
        for (int a = 0; a < 8; ++a) {
            float p = q0 * k0[a] + q1 * k1[a];
            #pragma unroll
            for (int off = 16; off > 0; off >>= 1) p += __shfl_xor(p, off, 32);
            s[a] = p;
        }
        float m = s[0];
        #pragma unroll
        for (int a = 1; a < 8; ++a) m = fmaxf(m, s[a]);
        float sum = 0.0f;
        #pragma unroll
        for (int a = 0; a < 8; ++a) { s[a] = __expf(s[a] - m); sum += s[a]; }
        float inv = 1.0f / sum;

        float o0 = 0.0f, o1 = 0.0f;
        #pragma unroll
        for (int a = 0; a < 8; ++a) {
            o0 += s[a] * bf2f(Vr[a * HD + d0]);
            o1 += s[a] * bf2f(Vr[a * HD + d0 + 1]);
        }
        o0 *= inv; o1 *= inv;

        #pragma unroll
        for (int rep = 0; rep < 8; ++rep) {
            size_t row = (size_t)(rep * NGRP + g);
            float2* dst = (float2*)&out1[row * HD + d0];
            *dst = make_float2(o0, o1);
        }
    }
}

// ---------------------------------------------------------------------------
// Host launch
// ---------------------------------------------------------------------------
extern "C" void kernel_launch(void* const* d_in, const int* in_sizes, int n_in,
                              void* d_out, int out_size, void* d_ws, size_t ws_size,
                              hipStream_t stream) {
    const float* self_obs = (const float*)d_in[0];
    const float* obs      = (const float*)d_in[1];
    const float* W_e1 = (const float*)d_in[2];  const float* b_e1 = (const float*)d_in[3];
    const float* W_e2 = (const float*)d_in[4];  const float* b_e2 = (const float*)d_in[5];
    const float* W_n1 = (const float*)d_in[6];  const float* b_n1 = (const float*)d_in[7];
    const float* W_n2 = (const float*)d_in[8];  const float* b_n2 = (const float*)d_in[9];
    const float* W_a1 = (const float*)d_in[10]; const float* b_a1 = (const float*)d_in[11];
    const float* W_a2 = (const float*)d_in[12]; const float* b_a2 = (const float*)d_in[13];

    char* wsb = (char*)d_ws;
    size_t off = 0;
    auto give = [&](size_t bytes) -> char* {
        char* p = wsb + off;
        off = (off + bytes + 255) & ~(size_t)255;
        return p;
    };
    unsigned short* wBe1 = (unsigned short*)give((size_t)32  * 512 * 2);
    unsigned short* wBe2 = (unsigned short*)give((size_t)512 * 512 * 2);
    unsigned short* wBn1 = (unsigned short*)give((size_t)512 * 512 * 2);
    unsigned short* wBn2 = (unsigned short*)give((size_t)512 * 512 * 2);
    unsigned short* wBa1 = (unsigned short*)give((size_t)512 * 512 * 2);
    unsigned short* wBa2 = (unsigned short*)give((size_t)512 * 512 * 2);
    unsigned short* X      = (unsigned short*)give((size_t)MROWS * 32 * 2);
    unsigned short* Ebuf   = (unsigned short*)give((size_t)MROWS * HD * 2);
    unsigned short* Vbuf   = (unsigned short*)give((size_t)MROWS * HD * 2);
    unsigned short* Aatt16 = (unsigned short*)give((size_t)BSZ * HD * 2);
    unsigned short* Vals   = (unsigned short*)give((size_t)BSZ * HD * 2);

    float* out1 = (float*)d_out;
    float* out2 = out1 + OUT1_ELEMS;

    pack_B_kernel<<<(32 * 512 + 255) / 256, 256, 0, stream>>>(W_e1, wBe1, 24, 32 * 512);
    pack_B_kernel<<<(512 * 512 + 255) / 256, 256, 0, stream>>>(W_e2, wBe2, 512, 512 * 512);
    pack_B_kernel<<<(512 * 512 + 255) / 256, 256, 0, stream>>>(W_n1, wBn1, 512, 512 * 512);
    pack_B_kernel<<<(512 * 512 + 255) / 256, 256, 0, stream>>>(W_n2, wBn2, 512, 512 * 512);
    pack_B_kernel<<<(512 * 512 + 255) / 256, 256, 0, stream>>>(W_a1, wBa1, 512, 512 * 512);
    pack_B_kernel<<<(512 * 512 + 255) / 256, 256, 0, stream>>>(W_a2, wBa2, 512, 512 * 512);

    build_X_kernel<<<(MROWS * 32 + 255) / 256, 256, 0, stream>>>(self_obs, obs, X);

    mlp2_bf16<<<MROWS / 64, 256, 0, stream>>>(X, 32, wBe1, b_e1, wBe2, b_e2, Ebuf);
    mlp2_bf16<<<MROWS / 64, 256, 0, stream>>>(Ebuf, 512, wBn1, b_n1, wBn2, b_n2, Vbuf);

    agent_attention_kernel<<<BSZ / 8, 256, 0, stream>>>(Ebuf, Vbuf, out2, Aatt16);

    mlp2_bf16<<<BSZ / 64, 256, 0, stream>>>(Aatt16, 512, wBa1, b_a1, wBa2, b_a2, Vals);

    multihead_kernel<<<NGRP / 8, 256, 0, stream>>>(out2, Vals, out1);
}
